// GCN_67886253081265
// MI455X (gfx1250) — matrix operand order repaired and checked by reading, not compiled, not run on hardware
//
#include <hip/hip_runtime.h>

// ---------------- types ----------------
typedef __attribute__((ext_vector_type(16))) __bf16        v16bf;
typedef __attribute__((ext_vector_type(8)))  float         v8f;
typedef __attribute__((ext_vector_type(4)))  unsigned int  u32x4;
typedef __attribute__((ext_vector_type(8)))  int           i32x8;
typedef __attribute__((ext_vector_type(4)))  int           i32x4;

#define NFEAT 165
#define KPAD1 192
#define HID   128

__device__ __forceinline__ unsigned short f2bf(float f) {
    union { float f; unsigned u; } v; v.f = f;
    unsigned r = v.u + 0x7FFFu + ((v.u >> 16) & 1u);   // round-to-nearest-even
    return (unsigned short)(r >> 16);
}

// ---------------- degree / norm ----------------
__global__ void k_fill1(float* p, int n) {
    int i = blockIdx.x * blockDim.x + threadIdx.x;
    if (i < n) p[i] = 1.0f;                       // self-loop contribution
}

__global__ void k_deg(const long long* __restrict__ dst, float* __restrict__ deg, long long E) {
    long long e = (long long)blockIdx.x * blockDim.x + threadIdx.x;
    if (e < E) atomicAdd(&deg[dst[e]], 1.0f);
}

__global__ void k_rsqrt(float* p, int n) {
    int i = blockIdx.x * blockDim.x + threadIdx.x;
    if (i < n) p[i] = rsqrtf(p[i]);               // deg >= 1 always (self-loops)
}

// ---------------- layernorm (one wave32 per node) -> bf16, K padded to 192 ----------------
__global__ void k_layernorm(const float* __restrict__ x,
                            const float* __restrict__ w, const float* __restrict__ b,
                            unsigned short* __restrict__ out, int N) {
    int lane = threadIdx.x & 31;
    int node = blockIdx.x * 8 + (threadIdx.x >> 5);
    if (node >= N) return;
    const float* xr = x + (size_t)node * NFEAT;
    float xv[6];
    float s = 0.f, s2 = 0.f;
    int cnt = 0;
    for (int c = lane; c < NFEAT; c += 32) {
        float v = xr[c];
        xv[cnt++] = v; s += v; s2 += v * v;
    }
    #pragma unroll
    for (int o = 16; o > 0; o >>= 1) {
        s  += __shfl_xor(s,  o, 32);
        s2 += __shfl_xor(s2, o, 32);
    }
    float mu   = s * (1.0f / NFEAT);
    float var  = s2 * (1.0f / NFEAT) - mu * mu;
    float rstd = rsqrtf(var + 1e-5f);
    unsigned short* orow = out + (size_t)node * KPAD1;
    cnt = 0;
    for (int c = lane; c < NFEAT; c += 32) {
        float y = (xv[cnt++] - mu) * rstd * w[c] + b[c];
        orow[c] = f2bf(y);
    }
    for (int c = NFEAT + lane; c < KPAD1; c += 32) orow[c] = 0;   // K padding
}

// ---------------- weight convert: W[K x 128] f32 -> Wt[128 x Kpad] bf16 (transposed, padded) ----
__global__ void k_convW(const float* __restrict__ W, unsigned short* __restrict__ Wt,
                        int K, int kpad) {
    int i = blockIdx.x * blockDim.x + threadIdx.x;
    int total = HID * kpad;
    if (i >= total) return;
    int n = i / kpad, k = i - n * kpad;
    Wt[(size_t)n * kpad + k] = (k < K) ? f2bf(W[(size_t)k * HID + n]) : (unsigned short)0;
}

// ---------------- WMMA GEMM: C[M x 128] = A[M x kpad](bf16) @ Wt'(bf16) ----------------
// block = 256 thr = 8 waves; each wave: 16-row strip x 128 cols (8 wmma accumulators).
// Weight operand staged to LDS once per block via the Tensor Data Mover; fragments
// read as b128 (16B-aligned by construction).
union Frag {
    unsigned int u[8];
    uint4        q[2];
    v16bf        v;
};

__global__ void __launch_bounds__(256)
k_gemm(const unsigned short* __restrict__ A, int lda,
       const unsigned short* __restrict__ Wt, int kpad,
       float* __restrict__ C, int M) {
    __shared__ unsigned short sW[HID * KPAD1];   // 48KB max (kpad<=192)

#if defined(__AMDGCN__) && __has_builtin(__builtin_amdgcn_tensor_load_to_lds)
    // --- TDM: DMA the whole 128 x kpad bf16 weight tile into LDS (wave 0 issues) ---
    if ((threadIdx.x >> 5) == 0) {
        unsigned           lds_off = (unsigned)(unsigned long long)(&sW[0]); // low 32b = LDS offset
        unsigned long long ga      = (unsigned long long)Wt;
        // D# group 0: count=1, lds_addr, global_addr[56:0], type=2 ("image")
        u32x4 g0 = { 1u,
                     lds_off,
                     (unsigned)(ga & 0xFFFFFFFFu),
                     (unsigned)(((ga >> 32) & 0x01FFFFFFu) | 0x80000000u) };
        // D# group 1: wg_mask=0, data_size=1 (2B), 2-D tile: dim0=kpad, dim1=HID rows
        i32x8 g1 = { (int)0x00010000u,                               // data_size=1
                     (int)((unsigned)(kpad & 0xFFFF) << 16),          // tensor_dim0[15:0]
                     (int)(((unsigned)kpad >> 16) | ((unsigned)HID << 16)), // td0 hi | tensor_dim1 lo
                     (int)((unsigned)(kpad & 0xFFFF) << 16),          // td1 hi(0) | tile_dim0=kpad
                     (int)HID,                                        // tile_dim1=HID, tile_dim2=0
                     (int)kpad,                                       // tensor_dim0_stride lo
                     0, 0 };
        i32x4 gz = { 0, 0, 0, 0 };
#if __clang_major__ >= 23
        i32x8 gz8 = { 0, 0, 0, 0, 0, 0, 0, 0 };
        __builtin_amdgcn_tensor_load_to_lds(g0, g1, gz, gz, gz8, 0);
#else
        __builtin_amdgcn_tensor_load_to_lds(g0, g1, gz, gz, 0);
#endif
        __builtin_amdgcn_s_wait_tensorcnt(0);
    }
    __syncthreads();
#else
    // fallback: cooperative copy
    for (int i = threadIdx.x; i < (HID * kpad) / 2; i += 256)
        ((unsigned int*)sW)[i] = ((const unsigned int*)Wt)[i];
    __syncthreads();
#endif

    int lane = threadIdx.x & 31;
    int wave = threadIdx.x >> 5;
    int hi   = lane >> 4;                 // half-wave select
    int m    = lane & 15;
    int rowBase = blockIdx.x * 128 + wave * 16;
    int r = rowBase + m;
    bool valid = r < M;
    const unsigned short* Arow = A + (size_t)(valid ? r : 0) * lda;

    v8f acc[8] = {};
    int khalfA = hi * 8;                  // elements
    int khalfB = hi * 16;                 // elements
    int n0 = lane & 15;

    for (int k0 = 0; k0 < kpad; k0 += 32) {
        if (k0 + 32 < kpad)
            __builtin_prefetch(Arow + k0 + 32, 0, 0);  // global_prefetch_b8
        Frag a;
        if (valid) {
            // ISA 16-bit A layout: q[0] <- K[k0+khalfA .. +7], q[1] <- K[k0+16+khalfA .. +7]
            a.q[0] = *(const uint4*)(Arow + k0 + khalfA);
            a.q[1] = *(const uint4*)(Arow + k0 + 16 + khalfA);
        } else {
            a.q[0] = uint4{0, 0, 0, 0};
            a.q[1] = uint4{0, 0, 0, 0};
        }
        #pragma unroll
        for (int j = 0; j < 8; ++j) {
            Frag bfrag;
            // ISA 16-bit B layout: column n0, q[0] <- K[k0+khalfA .. +7], q[1] <- K[k0+16+khalfA .. +7] (16B aligned)
            const uint4* Wq =
                (const uint4*)(sW + (size_t)(j * 16 + n0) * kpad + k0 + khalfA);
            bfrag.q[0] = Wq[0];                       // ds_load_b128
            bfrag.q[1] = Wq[2];                       // ds_load_b128
            acc[j] = __builtin_amdgcn_wmma_f32_16x16x32_bf16(
                false, a.v, false, bfrag.v, (short)0, acc[j], false, false);
        }
    }
    // C/D layout: vgpr rr -> row (hi*8 + rr), lane&15 -> col
    #pragma unroll
    for (int j = 0; j < 8; ++j) {
        int col = j * 16 + (lane & 15);
        #pragma unroll
        for (int rr = 0; rr < 8; ++rr) {
            int row = rowBase + hi * 8 + rr;
            if (row < M) C[(size_t)row * HID + col] = acc[j][rr];
        }
    }
}

// ---------------- zero ----------------
__global__ void k_zero(float* p, long long n) {
    long long i = (long long)blockIdx.x * blockDim.x + threadIdx.x;
    if (i < n) p[i] = 0.0f;
}

// ---------------- edge scatter: one wave per edge, 128 feats ----------------
__global__ void k_edge(const long long* __restrict__ src, const long long* __restrict__ dst,
                       const float* __restrict__ dinv,
                       const float* __restrict__ T, float* __restrict__ G, long long E) {
    long long gid = (long long)blockIdx.x * blockDim.x + threadIdx.x;
    long long e = gid >> 5;
    int lane = (int)(gid & 31);
    if (e >= E) return;
    long long s = src[e], d = dst[e];
    float nrm = dinv[s] * dinv[d];
    const float* ts = T + (size_t)s * HID;
    float* gd = G + (size_t)d * HID;
    #pragma unroll
    for (int i = 0; i < 4; ++i) {
        int c = lane + i * 32;
        atomicAdd(&gd[c], ts[c] * nrm);
    }
}

// ---------------- post: self-loop + bias (+relu), emit bf16 next-layer input / f32 out ----
__global__ void k_post(const float* __restrict__ T, const float* __restrict__ G,
                       const float* __restrict__ dinv, const float* __restrict__ bias,
                       int relu, unsigned short* bf16out, float* f32out, int N) {
    long long i = (long long)blockIdx.x * blockDim.x + threadIdx.x;
    if (i >= (long long)N * HID) return;
    int n = (int)(i >> 7);
    int c = (int)(i & 127);
    float di = dinv[n];
    float v = G[i] + T[i] * di * di + bias[c];
    if (relu) v = fmaxf(v, 0.0f);
    if (bf16out) bf16out[(size_t)n * HID + c] = f2bf(v);
    if (f32out)  f32out[i] = v;
}

// ---------------- output head: out = h3 @ Wout(128x2) + bout ----------------
__global__ void k_head(const float* __restrict__ h, const float* __restrict__ Wout,
                       const float* __restrict__ bout, float* __restrict__ out, int N) {
    int n = blockIdx.x * blockDim.x + threadIdx.x;
    if (n >= N) return;
    const float* hr = h + (size_t)n * HID;
    float a0 = bout[0], a1 = bout[1];
    #pragma unroll 4
    for (int c = 0; c < HID; ++c) {
        float v = hr[c];
        a0 += v * Wout[2 * c];
        a1 += v * Wout[2 * c + 1];
    }
    out[2 * n]     = a0;
    out[2 * n + 1] = a1;
}

// ---------------- host orchestration ----------------
extern "C" void kernel_launch(void* const* d_in, const int* in_sizes, int n_in,
                              void* d_out, int out_size, void* d_ws, size_t ws_size,
                              hipStream_t stream) {
    const float*     x    = (const float*)d_in[0];
    const long long* ei   = (const long long*)d_in[1];
    const float*     ln_w = (const float*)d_in[2];
    const float*     ln_b = (const float*)d_in[3];
    const float*     W1   = (const float*)d_in[4];
    const float*     b1   = (const float*)d_in[5];
    const float*     Wh   = (const float*)d_in[6];
    const float*     bh   = (const float*)d_in[7];
    const float*     W2   = (const float*)d_in[8];
    const float*     b2   = (const float*)d_in[9];
    const float*     Wout = (const float*)d_in[10];
    const float*     bout = (const float*)d_in[11];

    const int       N = in_sizes[0] / NFEAT;
    const long long E = in_sizes[1] / 2;
    const long long* src = ei;
    const long long* dst = ei + E;

    // workspace layout
    char* ws = (char*)d_ws;
    size_t off = 0;
    auto take = [&](size_t bytes) {
        off = (off + 255) & ~(size_t)255;
        size_t o = off; off += bytes; return o;
    };
    float*          dinv = (float*)         (ws + take((size_t)N * 4));
    unsigned short* Abf  = (unsigned short*)(ws + take((size_t)N * KPAD1 * 2));
    float*          T    = (float*)         (ws + take((size_t)N * HID * 4));
    float*          G    = (float*)         (ws + take((size_t)N * HID * 4));
    unsigned short* Wt   = (unsigned short*)(ws + take((size_t)HID * KPAD1 * 2));
    (void)ws_size; (void)n_in;

    float* out  = (float*)d_out;       // [N,2]
    float* hout = out + (size_t)2 * N; // [N,128]
    (void)out_size;

    const long long NH = (long long)N * HID;
    dim3 blk(256);
    int  gN    = (N + 255) / 256;
    int  gE    = (int)((E + 255) / 256);
    int  gEdge = (int)((E * 32 + 255) / 256);
    int  gNH   = (int)((NH + 255) / 256);
    int  gLN   = (N + 7) / 8;
    int  gGemm = (N + 127) / 128;
    int  gW1   = (HID * KPAD1 + 255) / 256;
    int  gWh   = (HID * HID + 255) / 256;

    // symmetric normalization
    k_fill1<<<gN, blk, 0, stream>>>(dinv, N);
    k_deg  <<<gE, blk, 0, stream>>>(dst, dinv, E);
    k_rsqrt<<<gN, blk, 0, stream>>>(dinv, N);

    // layernorm -> bf16 (padded K=192)
    k_layernorm<<<gLN, blk, 0, stream>>>(x, ln_w, ln_b, Abf, N);

    // layer 1: 165 -> 128, relu
    k_convW<<<gW1, blk, 0, stream>>>(W1, Wt, NFEAT, KPAD1);
    k_gemm <<<gGemm, blk, 0, stream>>>(Abf, KPAD1, Wt, KPAD1, T, N);
    k_zero <<<gNH, blk, 0, stream>>>(G, NH);
    k_edge <<<gEdge, blk, 0, stream>>>(src, dst, dinv, T, G, E);
    k_post <<<gNH, blk, 0, stream>>>(T, G, dinv, b1, 1, Abf, nullptr, N);

    // layer 2: 128 -> 128, relu
    k_convW<<<gWh, blk, 0, stream>>>(Wh, Wt, HID, HID);
    k_gemm <<<gGemm, blk, 0, stream>>>(Abf, HID, Wt, HID, T, N);
    k_zero <<<gNH, blk, 0, stream>>>(G, NH);
    k_edge <<<gEdge, blk, 0, stream>>>(src, dst, dinv, T, G, E);
    k_post <<<gNH, blk, 0, stream>>>(T, G, dinv, bh, 1, Abf, nullptr, N);

    // layer 3: 128 -> 128, no relu, f32 out (second tuple element)
    k_convW<<<gWh, blk, 0, stream>>>(W2, Wt, HID, HID);
    k_gemm <<<gGemm, blk, 0, stream>>>(Abf, HID, Wt, HID, T, N);
    k_zero <<<gNH, blk, 0, stream>>>(G, NH);
    k_edge <<<gEdge, blk, 0, stream>>>(src, dst, dinv, T, G, E);
    k_post <<<gNH, blk, 0, stream>>>(T, G, dinv, b2, 0, nullptr, hout, N);

    // output head
    k_head<<<gN, blk, 0, stream>>>(hout, Wout, bout, out, N);
}
